// LSTM_DOUBLE_ATT_STACK_PARALLEL_DROPOUT_ITEM_57990648430691
// MI455X (gfx1250) — matrix operand order repaired and checked by reading, not compiled, not run on hardware
//
#include <hip/hip_runtime.h>
#include <hip/hip_bf16.h>
#include <math.h>

#define B_    1024
#define R_    512
#define A_    196
#define L_    2
#define P_    4
#define BS    (B_ * A_)      /* 200704 rows of att viewed as 2-D */
#define GATES 8192           /* P * 4R */
#define K3    1536           /* 3R concatenated inputs */
#define BR    (B_ * R_)

typedef __attribute__((ext_vector_type(16))) __bf16 v16bf;
typedef __attribute__((ext_vector_type(8)))  __bf16 v8bf;
typedef __attribute__((ext_vector_type(8)))  float  v8f;

__device__ __forceinline__ __bf16 f2bf(float f) { return (__bf16)f; }

// Convert 8 consecutive fp32 into vector elements [base, base+8)
__device__ __forceinline__ void cvt8(const float* p, v16bf& r, int base) {
    const float4* q = (const float4*)p;
    float4 x0 = q[0], x1 = q[1];
    r[base + 0] = f2bf(x0.x); r[base + 1] = f2bf(x0.y);
    r[base + 2] = f2bf(x0.z); r[base + 3] = f2bf(x0.w);
    r[base + 4] = f2bf(x1.x); r[base + 5] = f2bf(x1.y);
    r[base + 6] = f2bf(x1.z); r[base + 7] = f2bf(x1.w);
}

// ---------------------------------------------------------------------------
// Weight prep: Wcat[n][k] bf16 with k = [Wi2h | Wh2h | Wa2h] rows, n = p*2048+g.
// Also bsum[n] = bi2h+bh2h+ba2h.
// ---------------------------------------------------------------------------
__global__ void __launch_bounds__(256)
prep_wcat_kernel(const float* __restrict__ Wi, const float* __restrict__ Wh,
                 const float* __restrict__ Wa, const float* __restrict__ bi,
                 const float* __restrict__ bh, const float* __restrict__ ba,
                 __bf16* __restrict__ Wc, float* __restrict__ bsum)
{
    int n = blockIdx.x;                    // 0..8191
    size_t src = (size_t)n * R_;           // [P,4R,R] flattens to [8192,512]
    __bf16* out = Wc + (size_t)n * K3;
    for (int k = threadIdx.x; k < R_; k += 256) {
        out[k]          = f2bf(Wi[src + k]);
        out[R_ + k]     = f2bf(Wh[src + k]);
        out[2 * R_ + k] = f2bf(Wa[src + k]);
    }
    if (threadIdx.x == 0) bsum[n] = bi[n] + bh[n] + ba[n];
}

// ---------------------------------------------------------------------------
// ah[b,a] = h[b,:] . W[a,:] + bias[a]   (small projection, VALU)
// ---------------------------------------------------------------------------
__global__ void __launch_bounds__(256)
proj_h_kernel(const float* __restrict__ h, const float* __restrict__ W,
              const float* __restrict__ bias, float* __restrict__ out)
{
    __shared__ float hs[R_];
    int b = blockIdx.x;
    for (int k = threadIdx.x; k < R_; k += 256) hs[k] = h[(size_t)b * R_ + k];
    __syncthreads();
    int a = threadIdx.x;
    if (a < A_) {
        const float* w = W + (size_t)a * R_;
        float acc = 0.f;
        #pragma unroll 8
        for (int k = 0; k < R_; ++k) acc += hs[k] * w[k];
        out[(size_t)b * A_ + a] = acc + bias[a];
    }
}

// ---------------------------------------------------------------------------
// Fused: score[b,s] = sum_a tanh( (att2d @ Wa^T)[row,a] + ba[a] + ah[row] ) * Wd[a] + bd
// One wave per 16-row M-tile; all 13 N-tiles (A padded 196->208) kept in regs.
// WMMA bf16, fp32 accumulate, epilogue reduced across lanes with shfl_xor.
// OOB columns of tile 12 use a clamped (in-bounds) row; their accumulator
// values are dead because the epilogue masks a >= 196 — no divergence in the
// hot loop, EXEC stays all-ones for every WMMA.
// ---------------------------------------------------------------------------
__global__ void __launch_bounds__(256)
attn_score_kernel(const float* __restrict__ att, const float* __restrict__ Wa,
                  const float* __restrict__ ba,  const float* __restrict__ Wd,
                  const float* __restrict__ bd,  const float* __restrict__ ah,
                  float* __restrict__ score)
{
    int wave = (blockIdx.x << 3) + (threadIdx.x >> 5);
    int lane = threadIdx.x & 31;
    int row0 = wave << 4;                          // exact: BS = 12544*16

    v8f z = {0.f, 0.f, 0.f, 0.f, 0.f, 0.f, 0.f, 0.f};
    v8f acc[13];
    #pragma unroll
    for (int t = 0; t < 13; ++t) acc[t] = z;

    int m    = lane & 15;
    int kb8  = (lane >> 4) << 3;
    int nl   = lane & 15;
    int kb16 = (lane >> 4) << 4;
    const float* arow = att + (size_t)(row0 + m) * R_;

    // Per-tile B row base, clamped in-bounds (branchless).
    const float* brow[13];
    #pragma unroll
    for (int t = 0; t < 13; ++t) {
        int n = (t << 4) + nl;
        n = (n < A_) ? n : (A_ - 1);
        brow[t] = Wa + (size_t)n * R_ + kb16;
    }

    for (int k0 = 0; k0 < R_; k0 += 32) {
        v16bf af;
        cvt8(arow + k0 + kb16,     af, 0);
        cvt8(arow + k0 + kb16 + 8, af, 8);
        #pragma unroll
        for (int t = 0; t < 13; ++t) {
            v16bf bf;
            cvt8(brow[t] + k0,     bf, 0);
            cvt8(brow[t] + k0 + 8, bf, 8);
            acc[t] = __builtin_amdgcn_wmma_f32_16x16x32_bf16(
                false, af, false, bf, (short)0, acc[t], false, false);
        }
    }

    // Epilogue: lane holds col n = nl, rows (lane>>4)*8 + j
    int mrow = (lane >> 4) << 3;
    float ahv[8];
    #pragma unroll
    for (int j = 0; j < 8; ++j) ahv[j] = ah[row0 + mrow + j];
    float ls[8];
    #pragma unroll
    for (int j = 0; j < 8; ++j) ls[j] = 0.f;
    #pragma unroll
    for (int t = 0; t < 13; ++t) {
        int a = (t << 4) + nl;
        if (a < A_) {                     // dead columns discarded here
            float wd  = Wd[a];
            float bav = ba[a];
            #pragma unroll
            for (int j = 0; j < 8; ++j)
                ls[j] += tanhf(acc[t][j] + bav + ahv[j]) * wd;
        }
    }
    #pragma unroll
    for (int off = 8; off >= 1; off >>= 1) {
        #pragma unroll
        for (int j = 0; j < 8; ++j) ls[j] += __shfl_xor(ls[j], off, 32);
    }
    if (nl == 0) {
        float b0 = bd[0];
        #pragma unroll
        for (int j = 0; j < 8; ++j) score[row0 + mrow + j] = ls[j] + b0;
    }
}

// ---------------------------------------------------------------------------
// Softmax over s (196) per batch row.
// ---------------------------------------------------------------------------
__global__ void __launch_bounds__(256)
softmax_kernel(const float* __restrict__ score, float* __restrict__ w)
{
    __shared__ float red[256];
    int b = blockIdx.x, t = threadIdx.x;
    float v = (t < A_) ? score[(size_t)b * A_ + t] : -3.0e38f;
    red[t] = v; __syncthreads();
    for (int s = 128; s > 0; s >>= 1) {
        if (t < s) red[t] = fmaxf(red[t], red[t + s]);
        __syncthreads();
    }
    float mx = red[0]; __syncthreads();
    float e = (t < A_) ? expf(v - mx) : 0.f;
    red[t] = e; __syncthreads();
    for (int s = 128; s > 0; s >>= 1) {
        if (t < s) red[t] += red[t + s];
        __syncthreads();
    }
    if (t < A_) w[(size_t)b * A_ + t] = e / red[0];
}

// ---------------------------------------------------------------------------
// att_res[b,r] = sum_s att[b,s,r] * w[b,s]  (fp32, coalesced over r)
// ---------------------------------------------------------------------------
__global__ void __launch_bounds__(256)
attres_kernel(const float* __restrict__ att, const float* __restrict__ w,
              float* __restrict__ out)
{
    __shared__ float ws[A_];
    int b = blockIdx.x, t = threadIdx.x;
    if (t < A_) ws[t] = w[(size_t)b * A_ + t];
    __syncthreads();
    const float* ab = att + (size_t)b * A_ * R_;
    for (int r = t; r < R_; r += 256) {
        float acc = 0.f;
        for (int s = 0; s < A_; ++s) acc += ab[(size_t)s * R_ + r] * ws[s];
        out[(size_t)b * R_ + r] = acc;
    }
}

// ---------------------------------------------------------------------------
// X3[b] = bf16 [ xt | prev_h | att_res ], xt = x (+ top_h_prev when has_add)
// ---------------------------------------------------------------------------
__global__ void __launch_bounds__(256)
build_x3_kernel(const float* __restrict__ x, const float* __restrict__ xadd,
                int has_add, const float* __restrict__ prev_h,
                const float* __restrict__ attres, __bf16* __restrict__ X3)
{
    int b = blockIdx.x;
    __bf16* out = X3 + (size_t)b * K3;
    for (int k = threadIdx.x; k < K3; k += 256) {
        float v;
        if (k < R_)            v = x[(size_t)b * R_ + k] +
                                   (has_add ? xadd[(size_t)b * R_ + k] : 0.f);
        else if (k < 2 * R_)   v = prev_h[(size_t)b * R_ + (k - R_)];
        else                   v = attres[(size_t)b * R_ + (k - 2 * R_)];
        out[k] = f2bf(v);
    }
}

// ---------------------------------------------------------------------------
// sums[b, n] = X3[b,:] @ Wcat[n,:]   M=1024 N=8192 K=1536, bf16 WMMA.
// Block = 8 waves sharing a 16-row tile; wave covers 128 columns (8 N-tiles).
// ---------------------------------------------------------------------------
__global__ void __launch_bounds__(256)
gates_gemm_kernel(const __bf16* __restrict__ X3, const __bf16* __restrict__ Wc,
                  float* __restrict__ sums)
{
    int wv   = threadIdx.x >> 5;
    int lane = threadIdx.x & 31;
    int row0 = blockIdx.y << 4;
    int col0 = (blockIdx.x << 10) + (wv << 7);

    v8f z = {0.f, 0.f, 0.f, 0.f, 0.f, 0.f, 0.f, 0.f};
    v8f acc[8];
    #pragma unroll
    for (int t = 0; t < 8; ++t) acc[t] = z;

    int m    = lane & 15;
    int kb8  = (lane >> 4) << 3;
    int nl   = lane & 15;
    int kb16 = (lane >> 4) << 4;
    const __bf16* arow = X3 + (size_t)(row0 + m) * K3;
    const __bf16* bbase = Wc + (size_t)(col0 + nl) * K3 + kb16;

    for (int k0 = 0; k0 < K3; k0 += 32) {
        v16bf af;
        {
            v8bf lo = *(const v8bf*)(arow + k0 + kb16);
            v8bf hi = *(const v8bf*)(arow + k0 + kb16 + 8);
            #pragma unroll
            for (int j = 0; j < 8; ++j) { af[j] = lo[j]; af[8 + j] = hi[j]; }
        }
        #pragma unroll
        for (int t = 0; t < 8; ++t) {
            const __bf16* brow = bbase + (size_t)(t << 4) * K3 + k0;
            v8bf b0 = *(const v8bf*)brow;
            v8bf b1 = *(const v8bf*)(brow + 8);
            v16bf bf;
            #pragma unroll
            for (int j = 0; j < 8; ++j) { bf[j] = b0[j]; bf[8 + j] = b1[j]; }
            acc[t] = __builtin_amdgcn_wmma_f32_16x16x32_bf16(
                false, af, false, bf, (short)0, acc[t], false, false);
        }
    }

    int mrow = (lane >> 4) << 3;
    #pragma unroll
    for (int t = 0; t < 8; ++t) {
        int n = col0 + (t << 4) + nl;
        #pragma unroll
        for (int j = 0; j < 8; ++j)
            sums[(size_t)(row0 + mrow + j) * GATES + n] = acc[t][j];
    }
}

// ---------------------------------------------------------------------------
// LSTM elementwise: gates per parallel p, nc, mean over P.
// ---------------------------------------------------------------------------
__global__ void __launch_bounds__(256)
lstm_kernel(const float* __restrict__ sums, const float* __restrict__ bsum,
            const float* __restrict__ prev_c, float* __restrict__ next_c,
            float* __restrict__ next_h)
{
    int b = blockIdx.x;
    for (int r = threadIdx.x; r < R_; r += 256) {
        float pc = prev_c[(size_t)b * R_ + r];
        float cs = 0.f, hs = 0.f;
        #pragma unroll
        for (int p = 0; p < P_; ++p) {
            int base = p * 2048;
            const float* s  = sums + (size_t)b * GATES + base;
            const float* bb = bsum + base;
            float si = s[r]            + bb[r];
            float sf = s[R_ + r]       + bb[R_ + r];
            float so = s[2 * R_ + r]   + bb[2 * R_ + r];
            float sg = s[3 * R_ + r]   + bb[3 * R_ + r];
            float ig = 1.f / (1.f + expf(-si));
            float fg = 1.f / (1.f + expf(-sf));
            float og = 1.f / (1.f + expf(-so));
            float it = tanhf(sg);
            float nc = fg * pc + ig * it;
            cs += nc;
            hs += og * tanhf(nc);
        }
        next_c[(size_t)b * R_ + r] = cs * 0.25f;
        next_h[(size_t)b * R_ + r] = hs * 0.25f;
    }
}

__global__ void __launch_bounds__(256)
add_kernel(const float* __restrict__ a, const float* __restrict__ b,
           float* __restrict__ o, int n)
{
    int i = blockIdx.x * 256 + threadIdx.x;
    if (i < n) o[i] = a[i] + b[i];
}

// ---------------------------------------------------------------------------
extern "C" void kernel_launch(void* const* d_in, const int* in_sizes, int n_in,
                              void* d_out, int out_size, void* d_ws, size_t ws_size,
                              hipStream_t stream)
{
    const float* x      = (const float*)d_in[0];
    const float* att    = (const float*)d_in[1];
    const float* inputs = (const float*)d_in[2];
    const float* Wa2a   = (const float*)d_in[3];
    const float* ba2a   = (const float*)d_in[4];
    const float* Wh2a   = (const float*)d_in[5];
    const float* bh2a   = (const float*)d_in[6];
    const float* Wd2d   = (const float*)d_in[7];
    const float* bd2d   = (const float*)d_in[8];
    const float* Wa2a1  = (const float*)d_in[9];
    const float* ba2a1  = (const float*)d_in[10];
    const float* Wh2a1  = (const float*)d_in[11];
    const float* bh2a1  = (const float*)d_in[12];
    const float* Wd2d1  = (const float*)d_in[13];
    const float* bd2d1  = (const float*)d_in[14];
    const float* Wi2h   = (const float*)d_in[15];
    const float* bi2h   = (const float*)d_in[16];
    const float* Wh2h   = (const float*)d_in[17];
    const float* bh2h   = (const float*)d_in[18];
    const float* Wa2h   = (const float*)d_in[19];
    const float* ba2h   = (const float*)d_in[20];
    float* out = (float*)d_out;

    char* ws = (char*)d_ws;
    auto alloc = [&](size_t bytes) -> char* {
        char* p = ws; ws += (bytes + 255) & ~(size_t)255; return p;
    };
    __bf16* Wcat   = (__bf16*)alloc((size_t)GATES * K3 * sizeof(__bf16)); // 25.2 MB
    float*  bsum   = (float*) alloc((size_t)GATES * sizeof(float));
    float*  ah     = (float*) alloc((size_t)BS * sizeof(float));
    float*  score  = (float*) alloc((size_t)BS * sizeof(float));
    float*  wsm    = (float*) alloc((size_t)BS * sizeof(float));
    float*  attres = (float*) alloc((size_t)BR * sizeof(float));
    float*  attre1 = (float*) alloc((size_t)BR * sizeof(float));
    __bf16* X3     = (__bf16*)alloc((size_t)B_ * K3 * sizeof(__bf16));
    float*  sums   = (float*) alloc((size_t)B_ * GATES * sizeof(float));  // 33.5 MB
    float*  hnext  = (float*) alloc((size_t)BR * sizeof(float));

    prep_wcat_kernel<<<GATES, 256, 0, stream>>>(Wi2h, Wh2h, Wa2h,
                                                bi2h, bh2h, ba2h, Wcat, bsum);

    const int score_blocks = BS / 16 / 8;  // 1568 (one wave per 16-row tile)

    for (int i = 0; i < L_; ++i) {
        const float* prev_c = inputs + (size_t)(2 * i) * BR;
        const float* prev_h = inputs + (size_t)(2 * i + 1) * BR;

        // ---- attention block 1 (driven by prev_h) ----
        proj_h_kernel<<<B_, 256, 0, stream>>>(prev_h, Wh2a, bh2a, ah);
        attn_score_kernel<<<score_blocks, 256, 0, stream>>>(att, Wa2a, ba2a,
                                                            Wd2d, bd2d, ah, score);
        softmax_kernel<<<B_, 256, 0, stream>>>(score, wsm);
        attres_kernel<<<B_, 256, 0, stream>>>(att, wsm, attres);

        // ---- fused gate GEMM ----
        build_x3_kernel<<<B_, 256, 0, stream>>>(x, out + BR, i, prev_h, attres, X3);
        gates_gemm_kernel<<<dim3(8, 64), 256, 0, stream>>>(X3, Wcat, sums);
        lstm_kernel<<<B_, 256, 0, stream>>>(sums, bsum, prev_c,
                                            out + (size_t)(2 * i) * BR, hnext);

        // ---- attention block 2 (driven by next_h) ----
        proj_h_kernel<<<B_, 256, 0, stream>>>(hnext, Wh2a1, bh2a1, ah);
        attn_score_kernel<<<score_blocks, 256, 0, stream>>>(att, Wa2a1, ba2a1,
                                                            Wd2d1, bd2d1, ah, score);
        softmax_kernel<<<B_, 256, 0, stream>>>(score, wsm);
        attres_kernel<<<B_, 256, 0, stream>>>(att, wsm, attre1);

        add_kernel<<<(BR + 255) / 256, 256, 0, stream>>>(attre1, hnext,
                                                         out + (size_t)(2 * i + 1) * BR, BR);
    }
}
